// CompositionalMLP_79001628442944
// MI455X (gfx1250) — matrix operand order repaired and checked by reading, not compiled, not run on hardware
//
#include <hip/hip_runtime.h>

// ---------------------------------------------------------------------------
// CompositionalMLP forward on gfx1250 (MI455X), wave32 + v_wmma_f32_16x16x32_f16
// ---------------------------------------------------------------------------

typedef _Float16 half_t;
typedef __attribute__((ext_vector_type(16))) _Float16 v16h;
typedef __attribute__((ext_vector_type(8)))  _Float16 v8h;
typedef __attribute__((ext_vector_type(8)))  float    v8f;

#define BATCH 32768
#define FDIM  32
#define HDIM  256
#define MMOD  4
#define ODIM  8
#define INW   144   // input_val row width (4*32 feats + 4*4 onehots)

// ---- fragment loaders -----------------------------------------------------
// A (16x32 f16): lane L: row = L&15, kgroup = L>>4. K = kg*16 .. +15 contiguous
// -> two 16B loads.
__device__ __forceinline__ v16h load_a_frag(const half_t* p_lo, const half_t* p_hi) {
  union { v16h v; v8h h[2]; } u;
  u.h[0] = *(const v8h*)p_lo;
  u.h[1] = *(const v8h*)p_hi;
  return u.v;
}
// B (32x16 f16): lane L: col = L&15, K = (L>>4)*16 .. +15 contiguous -> two 16B loads.
__device__ __forceinline__ v16h load_b_frag(const half_t* p) {
  union { v16h v; v8h h[2]; } u;
  u.h[0] = *(const v8h*)p;
  u.h[1] = *(const v8h*)(p + 8);
  return u.v;
}

// ---- prep kernels ---------------------------------------------------------
// f32 features -> f16 (first 128 cols of input_val), contiguous [B][128]
__global__ __launch_bounds__(256) void conv_feat(const float* __restrict__ in,
                                                 half_t* __restrict__ out) {
  long i = (long)blockIdx.x * blockDim.x + threadIdx.x;
  if (i < (long)BATCH * 128) {
    long b = i >> 7; int c = (int)(i & 127);
    out[i] = (half_t)in[b * INW + c];
  }
}

// transpose+convert: dst[m][o][i] (f16) = src[m][OFF+i][o] (f32)
// src row-count per module = IN_TOTAL, dst K-dim = IN, out dim = OUT.
__global__ __launch_bounds__(256) void conv_wt(const float* __restrict__ src,
                                               half_t* __restrict__ dst,
                                               int IN, int OUT, int IN_TOTAL,
                                               int OFF, int total) {
  int i = blockIdx.x * blockDim.x + threadIdx.x;
  if (i >= total) return;
  int ii = i % IN;
  int o  = (i / IN) % OUT;
  int m  = i / (IN * OUT);
  dst[i] = (half_t)src[((long)m * IN_TOTAL + OFF + ii) * OUT + o];
}

// ---- fused node kernel ----------------------------------------------------
// Block: 256 threads = 8 waves; covers 32 rows x 256 cols x all 4 modules.
// wave wid: row strip r0 = (wid>>2)*16, col group c0 = (wid&3)*64 (4 n-tiles).
template <bool HAS_PREV, bool FINAL>
__global__ __launch_bounds__(256) void node_kernel(
    const half_t* __restrict__ xprev,   // [B][256] f16
    const half_t* __restrict__ featH,   // [B][128] f16
    const float*  __restrict__ inputv,  // [B][144] f32 (onehots)
    int j,                              // node index (feat/onehot slice)
    const half_t* __restrict__ WpreT,   // [M][256][32]  f16 (transposed)
    const float*  __restrict__ bpre,    // [M][256]
    const half_t* __restrict__ WAT,     // [M][256][256] f16 (x_prev part)
    const half_t* __restrict__ WBT,     // [M][256][256] f16 (hp part)
    const float*  __restrict__ bpost,   // [M][256]
    const float*  __restrict__ Wfin,    // [M][256][8]  (FINAL)
    const float*  __restrict__ bfin,    // [M][8]       (FINAL)
    half_t* __restrict__ xout,          // [B][256] f16 (!FINAL)
    float*  __restrict__ outp)          // [B][8]       (FINAL)
{
  __shared__ __align__(16) half_t sX [32 * 256];
  __shared__ __align__(16) half_t sHp[32 * 256];
  __shared__ __align__(16) half_t sHq[FINAL ? 32 * 256 : 8];
  __shared__ float sOh[32][4];

  const int tid  = threadIdx.x;
  const int lane = tid & 31;
  const int wid  = tid >> 5;
  const int r0   = (wid >> 2) * 16;
  const int c0   = (wid & 3) * 64;
  const int ln   = lane & 15;
  const int kg   = lane >> 4;   // K-group for A frags; also row-half for C/D
  const long gr0 = (long)blockIdx.x * 32;

  // stage x_prev tile in LDS (reused 4m x 4n times)
  if (HAS_PREV) {
    const v8h* src = (const v8h*)(xprev + gr0 * 256);
    v8h* dst = (v8h*)sX;
    for (int i = tid; i < 32 * 256 / 8; i += 256) dst[i] = src[i];
  }
  if (tid < 128) {
    int r = tid >> 2, m = tid & 3;
    sOh[r][m] = inputv[(gr0 + r) * INW + 4 * FDIM + j * MMOD + m];
  }
  __syncthreads();

  v8f accX[4] = {};          // !FINAL: one-hot-selected x accumulator (f32)
  float outv = 0.f;          // FINAL: per-thread output accumulator
  const int frow = tid >> 3, fo = tid & 7;

  for (int m = 0; m < MMOD; ++m) {
    // ---- step 1: hp = relu(feat @ Wpre[m] + bpre[m]), K=32, -> LDS f16 ----
    {
      const half_t* arow = featH + (gr0 + r0 + ln) * 128 + j * FDIM + kg * 16;
      v16h a = load_a_frag(arow, arow + 8);
      for (int t = 0; t < 4; ++t) {
        int n0 = c0 + t * 16;
        const half_t* brow = WpreT + ((m * HDIM + n0 + ln) * FDIM) + kg * 16;
        v16h b = load_b_frag(brow);
        v8f c = {};
        c = __builtin_amdgcn_wmma_f32_16x16x32_f16(false, a, false, b,
                                                   (short)0, c, false, false);
        int col = n0 + ln;
        float bb = bpre[m * HDIM + col];
        for (int g = 0; g < 8; ++g) {
          float v = c[g] + bb;
          v = v > 0.f ? v : 0.f;
          sHp[(r0 + g + 8 * kg) * 256 + col] = (half_t)v;
        }
      }
    }
    __syncthreads();

    // ---- step 2: hq = relu(x_prev@WAT[m] + hp@WBT[m] + bpost[m]), K=256 ---
    v8f acc[4] = {};
    for (int kt = 0; kt < 8; ++kt) {
      int kb = kt * 32;
      const half_t* hrow = sHp + (r0 + ln) * 256 + kb + kg * 16;
      v16h ah = load_a_frag(hrow, hrow + 8);
      v16h ax = {};
      if constexpr (HAS_PREV) {
        const half_t* xrow = sX + (r0 + ln) * 256 + kb + kg * 16;
        ax = load_a_frag(xrow, xrow + 8);
      }
      for (int t = 0; t < 4; ++t) {
        int n0 = c0 + t * 16;
        const half_t* bB = WBT + ((m * HDIM + n0 + ln) * HDIM) + kb + kg * 16;
        v16h b1 = load_b_frag(bB);
        acc[t] = __builtin_amdgcn_wmma_f32_16x16x32_f16(false, ah, false, b1,
                                                        (short)0, acc[t], false, false);
        if constexpr (HAS_PREV) {
          const half_t* bA = WAT + ((m * HDIM + n0 + ln) * HDIM) + kb + kg * 16;
          v16h b2 = load_b_frag(bA);
          acc[t] = __builtin_amdgcn_wmma_f32_16x16x32_f16(false, ax, false, b2,
                                                          (short)0, acc[t], false, false);
        }
      }
    }

    // ---- bias + relu, then select-accumulate or stash for final linear ----
    for (int t = 0; t < 4; ++t) {
      int col = c0 + t * 16 + ln;
      float bb = bpost[m * HDIM + col];
      for (int g = 0; g < 8; ++g) {
        float v = acc[t][g] + bb;
        v = v > 0.f ? v : 0.f;
        if constexpr (FINAL) {
          sHq[(r0 + g + 8 * kg) * 256 + col] = (half_t)v;
        } else {
          accX[t][g] += sOh[r0 + g + 8 * kg][m] * v;
        }
      }
    }

    if constexpr (FINAL) {
      __syncthreads();
      // out[b] += oh[b,m] * (hq[b,m,:] @ Wfin[m] + bfin[m]); thread=(row,o)
      float dot = 0.f;
      for (int k = 0; k < HDIM; ++k)
        dot += (float)sHq[frow * 256 + k] * Wfin[(m * HDIM + k) * ODIM + fo];
      outv += sOh[frow][m] * (dot + bfin[m * ODIM + fo]);
    }
    __syncthreads();
  }

  if constexpr (FINAL) {
    outp[(gr0 + frow) * ODIM + fo] = outv;
  } else {
    for (int t = 0; t < 4; ++t) {
      int col = c0 + t * 16 + ln;
      for (int g = 0; g < 8; ++g) {
        long row = gr0 + r0 + g + 8 * kg;
        xout[row * 256 + col] = (half_t)accX[t][g];
      }
    }
  }
}

// ---------------------------------------------------------------------------
extern "C" void kernel_launch(void* const* d_in, const int* in_sizes, int n_in,
                              void* d_out, int out_size, void* d_ws, size_t ws_size,
                              hipStream_t stream) {
  (void)in_sizes; (void)n_in; (void)out_size; (void)ws_size;

  const float* input = (const float*)d_in[0];
  const float* W00   = (const float*)d_in[1];
  const float* b00   = (const float*)d_in[2];
  const float* W01   = (const float*)d_in[3];
  const float* b01   = (const float*)d_in[4];
  const float* Wpre[3]  = {(const float*)d_in[5],  (const float*)d_in[9],  (const float*)d_in[13]};
  const float* bpre[3]  = {(const float*)d_in[6],  (const float*)d_in[10], (const float*)d_in[14]};
  const float* Wpost[3] = {(const float*)d_in[7],  (const float*)d_in[11], (const float*)d_in[15]};
  const float* bpost[3] = {(const float*)d_in[8],  (const float*)d_in[12], (const float*)d_in[16]};
  const float* Wfin3 = (const float*)d_in[17];
  const float* bfin3 = (const float*)d_in[18];
  float* out = (float*)d_out;

  // carve workspace (256B aligned slices)
  size_t off = 0;
  auto carve = [&](size_t bytes) -> void* {
    off = (off + 255) & ~(size_t)255;
    void* p = (char*)d_ws + off;
    off += bytes;
    return p;
  };
  half_t* featH = (half_t*)carve((size_t)BATCH * 128 * 2);
  half_t* xA    = (half_t*)carve((size_t)BATCH * HDIM * 2);
  half_t* xB    = (half_t*)carve((size_t)BATCH * HDIM * 2);
  half_t* W00t  = (half_t*)carve((size_t)MMOD * HDIM * FDIM * 2);
  half_t* W01t  = (half_t*)carve((size_t)MMOD * HDIM * HDIM * 2);
  half_t *WpreT[3], *WAT[3], *WBT[3];
  for (int i = 0; i < 3; ++i) {
    WpreT[i] = (half_t*)carve((size_t)MMOD * HDIM * FDIM * 2);
    WAT[i]   = (half_t*)carve((size_t)MMOD * HDIM * HDIM * 2);
    WBT[i]   = (half_t*)carve((size_t)MMOD * HDIM * HDIM * 2);
  }

  // ---- prep: f16 conversions / transposes ----
  {
    int total = BATCH * 128;
    conv_feat<<<(total + 255) / 256, 256, 0, stream>>>(input, featH);
  }
  auto cw = [&](const float* src, half_t* dst, int IN, int OUT, int IN_TOTAL, int OFF) {
    int total = MMOD * IN * OUT;
    conv_wt<<<(total + 255) / 256, 256, 0, stream>>>(src, dst, IN, OUT, IN_TOTAL, OFF, total);
  };
  cw(W00, W00t, FDIM, HDIM, FDIM, 0);
  cw(W01, W01t, HDIM, HDIM, HDIM, 0);
  for (int i = 0; i < 3; ++i) {
    cw(Wpre[i],  WpreT[i], FDIM, HDIM, FDIM, 0);
    cw(Wpost[i], WAT[i],   HDIM, HDIM, 2 * HDIM, 0);     // Wpost[:, :H, :]
    cw(Wpost[i], WBT[i],   HDIM, HDIM, 2 * HDIM, HDIM);  // Wpost[:, H:, :]
  }

  // ---- node chain ----
  dim3 grid(BATCH / 32), block(256);
  // node 0: no prev, select -> xA
  node_kernel<false, false><<<grid, block, 0, stream>>>(
      nullptr, featH, input, 0, W00t, b00, nullptr, W01t, b01,
      nullptr, nullptr, xA, nullptr);
  // node 1: xA -> xB
  node_kernel<true, false><<<grid, block, 0, stream>>>(
      xA, featH, input, 1, WpreT[0], bpre[0], WAT[0], WBT[0], bpost[0],
      nullptr, nullptr, xB, nullptr);
  // node 2: xB -> xA
  node_kernel<true, false><<<grid, block, 0, stream>>>(
      xB, featH, input, 2, WpreT[1], bpre[1], WAT[1], WBT[1], bpost[1],
      nullptr, nullptr, xA, nullptr);
  // node 3: xA -> out (fused final linear + select)
  node_kernel<true, true><<<grid, block, 0, stream>>>(
      xA, featH, input, 3, WpreT[2], bpre[2], WAT[2], WBT[2], bpost[2],
      Wfin3, bfin3, nullptr, out);
}
